// MHSA_1391569404222
// MI455X (gfx1250) — matrix operand order repaired and checked by reading, not compiled, not run on hardware
//
#include <hip/hip_runtime.h>

// ---------------------------------------------------------------------------
// MHSA with per-head LoRA projection, optimized for gfx1250 (MI455X):
//   Wc[h] = Wa[h] @ Wt[h]            (12 x 64 x 768, bf16)
//   t[b,h] = X[b] @ Wc[h]^T          (WMMA bf16 -> f32)
//   attn   = softmax(t t^T * scale)  (flash-style, WMMA bf16 -> f32)
//   out[b,n,h*64+d] = attn @ t
// Data movement: async global->LDS (ASYNCcnt) with double-buffered tiles.
// ---------------------------------------------------------------------------

constexpr int Bv    = 8;
constexpr int Nv    = 1024;
constexpr int HIDv  = 768;
constexpr int HEADS = 12;
constexpr int LORA  = 256;
constexpr int HDIM  = 64;

typedef __attribute__((ext_vector_type(16))) __bf16          v16bf;
typedef __attribute__((ext_vector_type(16))) unsigned short  v16us;
typedef __attribute__((ext_vector_type(8)))  float           v8f;

__device__ __forceinline__ unsigned short f2bf(float f) {
    unsigned u = __builtin_bit_cast(unsigned, f);
    u += 0x7FFFu + ((u >> 16) & 1u);          // round-to-nearest-even
    return (unsigned short)(u >> 16);
}

__device__ __forceinline__ v16bf as_bf(v16us u) {
    return __builtin_bit_cast(v16bf, u);
}

__device__ __forceinline__ v8f wmma_bf16(v16bf a, v16bf b, v8f c) {
    return __builtin_amdgcn_wmma_f32_16x16x32_bf16(false, a, false, b,
                                                   (short)0, c, false, false);
}

// Async copy of 16 bytes global -> LDS (tracked by ASYNCcnt, no VGPR data).
// LDS byte offset = low 32 bits of the generic pointer (LDS aperture).
__device__ __forceinline__ void async_b128(unsigned short* lds,
                                           const unsigned short* g) {
    unsigned off = (unsigned)(unsigned long long)lds;
    asm volatile("global_load_async_to_lds_b128 %0, %1, off"
                 :: "v"(off), "v"(g) : "memory");
}
#define WAIT_ASYNC(n) asm volatile("s_wait_asynccnt " #n ::: "memory")

// ---------------------------------------------------------------------------
// Kernel 0: X fp32 -> bf16
// ---------------------------------------------------------------------------
__global__ __launch_bounds__(256) void convert_x(const float* __restrict__ X,
                                                 unsigned short* __restrict__ Xb,
                                                 int n) {
    int i = (blockIdx.x * 256 + threadIdx.x) * 4;
    if (i + 3 < n) {
        Xb[i + 0] = f2bf(X[i + 0]);
        Xb[i + 1] = f2bf(X[i + 1]);
        Xb[i + 2] = f2bf(X[i + 2]);
        Xb[i + 3] = f2bf(X[i + 3]);
    }
}

// ---------------------------------------------------------------------------
// Kernel 1: Wc[h][e][d] = sum_l Wa[h][e][l] * Wt[h][l][d]   (fp32 acc -> bf16)
// ---------------------------------------------------------------------------
__global__ __launch_bounds__(256) void build_wc(const float* __restrict__ Wa,
                                                const float* __restrict__ Wt,
                                                unsigned short* __restrict__ Wc) {
    int idx = blockIdx.x * 256 + threadIdx.x;      // h*(HDIM*HID) + e*HID + d
    int d = idx % HIDv;
    int e = (idx / HIDv) % HDIM;
    int h = idx / (HIDv * HDIM);
    const float* wa = Wa + (h * HDIM + e) * LORA;
    const float* wt = Wt + h * LORA * HIDv + d;
    float s = 0.f;
    #pragma unroll 8
    for (int l = 0; l < LORA; ++l) s += wa[l] * wt[l * HIDv];
    Wc[idx] = f2bf(s);
}

// ---------------------------------------------------------------------------
// Kernel 2: t[b,h] = X[b] @ Wc[h]^T      ([1024 x 768] x [768 x 64] -> bf16)
// grid (8 row-blocks, 12 heads, 8 batch), 256 threads = 8 wave32
// Double-buffered async staging; each wave: 16 rows x 64 cols = 4 C-tiles.
// ---------------------------------------------------------------------------
__global__ __launch_bounds__(256) void proj_kernel(const unsigned short* __restrict__ X,
                                                   const unsigned short* __restrict__ Wc,
                                                   unsigned short* __restrict__ T) {
    __shared__ __align__(16) unsigned short Xs[2][128 * 32];
    __shared__ __align__(16) unsigned short Ws[2][64 * 32];

    const int tid  = threadIdx.x;
    const int lane = tid & 31;
    const int wave = tid >> 5;
    const int h = blockIdx.y, b = blockIdx.z;
    const int rowBase = blockIdx.x * 128;
    const int m  = lane & 15;       // A row / B col / C col
    const int kh = lane >> 4;       // K-half selector

    const unsigned long long xbase = (unsigned long long)b * Nv * HIDv;
    const int xr = tid >> 1, xc = (tid & 1) * 16;   // X stage: 16 ushort / thread
    const int wr = tid >> 2, wc = (tid & 3) * 8;    // W stage:  8 ushort / thread

    auto stage = [&](int buf, int kk) {
        const unsigned short* gx = X + xbase +
            (unsigned long long)(rowBase + xr) * HIDv + kk + xc;
        async_b128(&Xs[buf][xr * 32 + xc], gx);          // 3 async instrs / wave
        async_b128(&Xs[buf][xr * 32 + xc + 8], gx + 8);
        async_b128(&Ws[buf][wr * 32 + wc],
                   Wc + ((unsigned long long)h * HDIM + wr) * HIDv + kk + wc);
    };

    v8f zero = {};
    v8f acc[4] = {zero, zero, zero, zero};

    stage(0, 0);                                     // prefetch first K-chunk
    const int NIT = HIDv / 32;                       // 24
    for (int it = 0; it < NIT; ++it) {
        __syncthreads();                             // buf[next] free to overwrite
        if (it + 1 < NIT) {
            stage((it + 1) & 1, (it + 1) * 32);      // prefetch next chunk
            WAIT_ASYNC(0x3);                         // current chunk landed
        } else {
            WAIT_ASYNC(0x0);
        }
        __syncthreads();                             // all waves' parts visible
        const unsigned short* xs = Xs[it & 1];
        const unsigned short* ws = Ws[it & 1];

        // A fragment: 16-bit A 16x32 layout (lane m holds row m; K split by kh)
        v16us au;
        const int xrow = wave * 16 + m;
        #pragma unroll
        for (int i = 0; i < 8; ++i) {
            au[i]     = xs[xrow * 32 + kh * 8 + i];
            au[i + 8] = xs[xrow * 32 + 16 + kh * 8 + i];
        }
        v16bf a = as_bf(au);

        #pragma unroll
        for (int c = 0; c < 4; ++c) {
            // B fragment: B[k][n] = Wc[e = c*16+n][d], element i <-> K = kh*8+(i&7)+(i>>3)*16
            v16us bu;
            #pragma unroll
            for (int i = 0; i < 16; ++i)
                bu[i] = ws[(c * 16 + m) * 32 + kh * 8 + (i & 7) + (i >> 3) * 16];
            acc[c] = wmma_bf16(a, as_bf(bu), acc[c]);
        }
    }

    // store t (bf16): C layout -> row = rowBase + wave*16 + r + kh*8, col = c*16+m
    unsigned long long tbase =
        ((unsigned long long)(b * HEADS + h) * Nv + rowBase + wave * 16 + kh * 8) * HDIM;
    #pragma unroll
    for (int c = 0; c < 4; ++c)
        #pragma unroll
        for (int r = 0; r < 8; ++r)
            T[tbase + (unsigned long long)r * HDIM + c * 16 + m] = f2bf(acc[c][r]);
}

// ---------------------------------------------------------------------------
// Kernel 3: fused attention per (b, h, 128-row block); flash-style streaming
// softmax over 32-key chunks, double-buffered async key staging.
// 256 threads = 8 wave32; each wave owns 16 query rows.
// ---------------------------------------------------------------------------
__global__ __launch_bounds__(256) void attn_kernel(const unsigned short* __restrict__ T,
                                                   const int* __restrict__ mask,
                                                   float* __restrict__ out) {
    __shared__ __align__(16) unsigned short Ks[2][32 * 72];     // key chunks (padded)
    __shared__ __align__(16) unsigned short Ps[8 * 16 * 34];    // per-wave P scratch

    const int tid  = threadIdx.x;
    const int lane = tid & 31;
    const int wave = tid >> 5;
    const int h = blockIdx.y, b = blockIdx.z;
    const int qBase = blockIdx.x * 128 + wave * 16;
    const int m  = lane & 15;
    const int kh = lane >> 4;
    const float scale = 0.03608439182435161f;   // 768^-0.5

    const unsigned long long thead = (unsigned long long)(b * HEADS + h) * Nv * HDIM;
    const int kr = tid >> 3, kc = (tid & 7) * 8;     // key stage: 8 ushort / thread

    auto stage = [&](int buf, int j) {               // 1 async instr / wave
        async_b128(&Ks[buf][kr * 72 + kc],
                   T + thead + (unsigned long long)(j + kr) * HDIM + kc);
    };

    // Q fragments (two A-frags covering d = 0..31 and 32..63)
    v16us qu0, qu1;
    {
        const unsigned short* qrow = T + thead + (unsigned long long)(qBase + m) * HDIM;
        #pragma unroll
        for (int i = 0; i < 8; ++i) {
            qu0[i]     = qrow[kh * 8 + i];
            qu0[i + 8] = qrow[16 + kh * 8 + i];
            qu1[i]     = qrow[32 + kh * 8 + i];
            qu1[i + 8] = qrow[48 + kh * 8 + i];
        }
    }
    v16bf qa0 = as_bf(qu0), qa1 = as_bf(qu1);

    v8f zero = {};
    v8f acc[4] = {zero, zero, zero, zero};
    float rowM[8], rowL[8];
    #pragma unroll
    for (int r = 0; r < 8; ++r) { rowM[r] = -1e30f; rowL[r] = 0.f; }

    const int* mrow = mask + b * Nv;

    stage(0, 0);                                     // prefetch first key chunk
    const int NIT = Nv / 32;                         // 32
    for (int it = 0; it < NIT; ++it) {
        const int j = it * 32;
        __syncthreads();                             // buf[next] free to overwrite
        if (it + 1 < NIT) {
            stage((it + 1) & 1, j + 32);             // prefetch next chunk
            WAIT_ASYNC(0x1);                         // current chunk landed
        } else {
            WAIT_ASYNC(0x0);
        }
        __syncthreads();                             // all waves' parts visible
        const unsigned short* ks = Ks[it & 1];

        // S = Q K^T : two 16x16 score tiles (keys j..j+15, j+16..j+31), K-dim 64
        v8f s0 = zero, s1 = zero;
        #pragma unroll
        for (int kb = 0; kb < 2; ++kb) {
            v16us b0u, b1u;
            #pragma unroll
            for (int i = 0; i < 16; ++i) {
                int d = kb * 32 + kh * 8 + (i & 7) + (i >> 3) * 16;
                b0u[i] = ks[m * 72 + d];            // B[k][n] = K[key=j+n][d]
                b1u[i] = ks[(16 + m) * 72 + d];
            }
            v16bf aq = (kb == 0) ? qa0 : qa1;
            s0 = wmma_bf16(aq, as_bf(b0u), s0);
            s1 = wmma_bf16(aq, as_bf(b1u), s1);
        }

        const int mv0 = mrow[j + m];
        const int mv1 = mrow[j + 16 + m];

        // online softmax: row reductions across 16-lane groups (C layout)
        #pragma unroll
        for (int r = 0; r < 8; ++r) {
            float v0 = s0[r] * scale; if (mv0 == 0) v0 = -1e30f;
            float v1 = s1[r] * scale; if (mv1 == 0) v1 = -1e30f;
            float cm = fmaxf(v0, v1);
            cm = fmaxf(cm, __shfl_xor(cm, 1));
            cm = fmaxf(cm, __shfl_xor(cm, 2));
            cm = fmaxf(cm, __shfl_xor(cm, 4));
            cm = fmaxf(cm, __shfl_xor(cm, 8));
            float nm    = fmaxf(rowM[r], cm);
            float alpha = __expf(rowM[r] - nm);
            float p0 = __expf(v0 - nm);
            float p1 = __expf(v1 - nm);
            float ls = p0 + p1;
            ls += __shfl_xor(ls, 1);
            ls += __shfl_xor(ls, 2);
            ls += __shfl_xor(ls, 4);
            ls += __shfl_xor(ls, 8);
            rowL[r] = rowL[r] * alpha + ls;
            rowM[r] = nm;
            #pragma unroll
            for (int c = 0; c < 4; ++c) acc[c][r] *= alpha;
            Ps[(wave * 16 + kh * 8 + r) * 34 + m]      = f2bf(p0);
            Ps[(wave * 16 + kh * 8 + r) * 34 + 16 + m] = f2bf(p1);
        }

        // intra-wave LDS write->read ordering for P redistribution
        asm volatile("s_wait_dscnt 0" ::: "memory");

        // P as A-fragment (16x32 over the 32 keys of this chunk)
        v16us pu;
        #pragma unroll
        for (int i = 0; i < 8; ++i) {
            pu[i]     = Ps[(wave * 16 + m) * 34 + kh * 8 + i];
            pu[i + 8] = Ps[(wave * 16 + m) * 34 + 16 + kh * 8 + i];
        }
        v16bf pa = as_bf(pu);

        // O += P V : V as B-fragment, B[k][n] = t[key=j+k][d = c*16+n]
        #pragma unroll
        for (int c = 0; c < 4; ++c) {
            v16us vu;
            #pragma unroll
            for (int i = 0; i < 16; ++i)
                vu[i] = ks[(kh * 8 + (i & 7) + (i >> 3) * 16) * 72 + c * 16 + m];
            acc[c] = wmma_bf16(pa, as_bf(vu), acc[c]);
        }
    }

    // epilogue: out[b, row, h*64 + col] = acc / L
    #pragma unroll
    for (int r = 0; r < 8; ++r) {
        float inv = rowL[r] > 0.f ? 1.f / rowL[r] : 0.f;
        int row = qBase + kh * 8 + r;
        float* o = out + ((unsigned long long)(b * Nv + row)) * (HEADS * HDIM) + h * HDIM;
        #pragma unroll
        for (int c = 0; c < 4; ++c)
            o[c * 16 + m] = acc[c][r] * inv;
    }
}

// ---------------------------------------------------------------------------
extern "C" void kernel_launch(void* const* d_in, const int* in_sizes, int n_in,
                              void* d_out, int out_size, void* d_ws, size_t ws_size,
                              hipStream_t stream) {
    const float* hs   = (const float*)d_in[0];   // [8,1024,768] f32
    const int*   mask = (const int*)  d_in[1];   // [8,1024] i32
    const float* Wt   = (const float*)d_in[2];   // [12,256,768] f32
    const float* Wa   = (const float*)d_in[3];   // [12,64,256] f32
    float* out = (float*)d_out;                  // [8,1024,768] f32

    unsigned short* Xbf = (unsigned short*)d_ws;                    // 12.6 MB
    unsigned short* Wc  = Xbf + (size_t)Bv * Nv * HIDv;             //  1.2 MB
    unsigned short* Tbf = Wc  + (size_t)HEADS * HDIM * HIDv;        // 12.6 MB

    const int nx = Bv * Nv * HIDv;
    convert_x<<<nx / (256 * 4), 256, 0, stream>>>(hs, Xbf, nx);
    build_wc<<<(HEADS * HDIM * HIDv) / 256, 256, 0, stream>>>(Wa, Wt, Wc);

    dim3 grid(Nv / 128, HEADS, Bv);
    proj_kernel<<<grid, 256, 0, stream>>>(Xbf, Wc, Tbf);
    attn_kernel<<<grid, 256, 0, stream>>>(Tbf, mask, out);
    (void)in_sizes; (void)n_in; (void)out_size; (void)ws_size;
}
